// CrossAttentionMultiHead_74354473828906
// MI455X (gfx1250) — compile-verified, matrix operand order checked by reading, not run on hardware
//
#include <hip/hip_runtime.h>
#include <hip/hip_bf16.h>

typedef __attribute__((ext_vector_type(16))) _Float16 v16h;
typedef __attribute__((ext_vector_type(8)))  _Float16 v8h;
typedef __attribute__((ext_vector_type(2)))  _Float16 v2h;
typedef __attribute__((ext_vector_type(8)))  float    v8f;

#define NHEADS 4
#define DH 128
#define CC 512
#define EE 768
#define LL 77
#define LP 96
#define SS 4096
#define BB 8

__device__ __forceinline__ v8f zero8() {
  v8f z;
#pragma unroll
  for (int i = 0; i < 8; ++i) z[i] = 0.0f;
  return z;
}

// Build a 16x32 f16 A/B fragment from two contiguous 8-half chunks.
// lane<16  : elements = K[0..7] ++ K[16..23]   (koff = 0)
// lane>=16 : elements = K[8..15] ++ K[24..31]  (koff = 8)
__device__ __forceinline__ v16h frag_h(const _Float16* lo, const _Float16* hi) {
  v8h a = *(const v8h*)lo;
  v8h b = *(const v8h*)hi;
  v16h r;
#pragma unroll
  for (int i = 0; i < 8; ++i) { r[i] = a[i]; r[i + 8] = b[i]; }
  return r;
}

__device__ __forceinline__ v16h frag_f32(const float* lo, const float* hi) {
  v16h r;
#pragma unroll
  for (int i = 0; i < 8; ++i) { r[i] = (_Float16)lo[i]; r[i + 8] = (_Float16)hi[i]; }
  return r;
}

__device__ __forceinline__ v16h frag_zero() {
  v16h r;
#pragma unroll
  for (int i = 0; i < 16; ++i) r[i] = (_Float16)0.0f;
  return r;
}

__device__ __forceinline__ v8f wmma16(v16h a, v16h b, v8f c) {
  return __builtin_amdgcn_wmma_f32_16x16x32_f16(false, a, false, b, (short)0, c,
                                                false, false);
}

// ---------------------------------------------------------------------------
// prep: Wq f32 -> f16 (one-time)
// ---------------------------------------------------------------------------
__global__ __launch_bounds__(256) void prep_kernel(const float* __restrict__ wq,
                                                   _Float16* __restrict__ wqh) {
  int i = blockIdx.x * 256 + threadIdx.x;
  if (i < CC * CC) wqh[i] = (_Float16)wq[i];
}

// ---------------------------------------------------------------------------
// kv: K[b][lp][c] (f16) = text_emb @ Wk^T + bk ; Vt[b][c][lp] = (text @ Wv^T + bv)^T
// one wave per 16x16 output tile. grid = B * 6 * 32
// ---------------------------------------------------------------------------
__global__ __launch_bounds__(32) void kv_kernel(
    const float* __restrict__ te, const float* __restrict__ Wk_w,
    const float* __restrict__ Wk_b, const float* __restrict__ Wv_w,
    const float* __restrict__ Wv_b, _Float16* __restrict__ Kbuf,
    _Float16* __restrict__ Vtbuf) {
  const int idx = blockIdx.x;
  const int b = idx / 192;
  const int lt = (idx % 192) / 32;
  const int nt = idx % 32;
  const int lane = threadIdx.x & 31;
  const int rowA = lane & 15;
  const int halfsel = lane >> 4;
  const int koff = halfsel * 8;
  const int l0 = lt * 16;
  const int n0 = nt * 16;

  const int arow = l0 + rowA;
  const float* av = te + ((size_t)b * LL + arow) * EE;
  const float* wkr = Wk_w + (size_t)(n0 + rowA) * EE;
  const float* wvr = Wv_w + (size_t)(n0 + rowA) * EE;

  v8f accK = zero8(), accV = zero8();
  for (int kt = 0; kt < 24; ++kt) {
    int k0 = kt * 32;
    v16h a = (arow < LL) ? frag_f32(av + k0 + koff, av + k0 + 16 + koff)
                         : frag_zero();
    v16h bk = frag_f32(wkr + k0 + koff, wkr + k0 + 16 + koff);
    v16h bv = frag_f32(wvr + k0 + koff, wvr + k0 + 16 + koff);
    accK = wmma16(a, bk, accK);
    accV = wmma16(a, bv, accV);
  }
  const float bk = Wk_b[n0 + rowA];
  const float bv = Wv_b[n0 + rowA];
#pragma unroll
  for (int r = 0; r < 8; ++r) {
    int ro = l0 + r + halfsel * 8;  // token index (padded rows become bias; masked later)
    int co = n0 + rowA;             // channel index
    Kbuf[((size_t)b * LP + ro) * CC + co] = (_Float16)(accK[r] + bk);
    Vtbuf[((size_t)b * CC + co) * LP + ro] = (_Float16)(accV[r] + bv);
  }
}

// ---------------------------------------------------------------------------
// attn: block = (b, 32-query tile); 4 waves = 4 heads; 2 m-tiles of 16 queries.
// Every Wq/K/V fragment load feeds two WMMAs (both m-tiles).
// LDS phases:
//   phase 1: [xs 33280][qs0 17408][ps0 13312]            (stage + Q-proj)
//   phase 2: [qs1 17408 | ps1 13312][qs0][ps0]           (qs1/ps1 alias dead xs)
//   phase 3: [outT 40960 ...]                            (writeout)
// ---------------------------------------------------------------------------
#define XS_BYTES (32 * (CC + 8) * 2)                 // 33280
#define QS_BYTES (NHEADS * 16 * (DH + 8) * 2)        // 17408
#define PS_BYTES (NHEADS * 16 * (LP + 8) * 2)        // 13312
#define SMEM_BYTES (XS_BYTES + QS_BYTES + PS_BYTES)  // 64000 (>= outT 40960)

__global__ __launch_bounds__(128) void attn_kernel(
    const float* __restrict__ x, const int* __restrict__ amask,
    const _Float16* __restrict__ Kbuf, const _Float16* __restrict__ Vtbuf,
    const _Float16* __restrict__ Wqh, const float* __restrict__ Wq_b,
    float* __restrict__ out, float* __restrict__ part) {
  __shared__ __attribute__((aligned(16))) char smem[SMEM_BYTES];
  _Float16(*xs)[CC + 8] = (_Float16(*)[CC + 8])smem;  // [32][520]
  _Float16(*qs1)[16][DH + 8] = (_Float16(*)[16][DH + 8])smem;  // aliases xs
  _Float16(*ps1)[16][LP + 8] = (_Float16(*)[16][LP + 8])(smem + QS_BYTES);
  _Float16(*qs0)[16][DH + 8] = (_Float16(*)[16][DH + 8])(smem + XS_BYTES);
  _Float16(*ps0)[16][LP + 8] =
      (_Float16(*)[16][LP + 8])(smem + XS_BYTES + QS_BYTES);
  float(*outT)[DH][20] = (float(*)[DH][20])smem;  // aliases after final barrier

  const int b = blockIdx.x >> 7;  // 128 32-query tiles per batch
  const int stp = blockIdx.x & 127;
  const int s0 = stp << 5;
  const int tid = threadIdx.x;
  const int wave = tid >> 5;
  const int lane = tid & 31;
  const int rowA = lane & 15;
  const int halfsel = lane >> 4;
  const int koffA = halfsel * 8;
  const int h = wave;

  // stage x tile: x is [B][C][S]; 32 queries -> full 128B lines per channel.
  {
    const float* xb = x + (size_t)b * CC * SS + s0;
#pragma unroll
    for (int i = 0; i < 2; ++i) {
      int c = 2 * tid + i * 256;
      const float* p0 = xb + (size_t)c * SS;
      const float* p1 = p0 + SS;
#pragma unroll
      for (int s = 0; s < 32; ++s) {
        v2h pk;
        pk[0] = (_Float16)p0[s];
        pk[1] = (_Float16)p1[s];
        *(v2h*)&xs[s][c] = pk;
      }
    }
  }
  __syncthreads();

  // ---- Q projection: both m-tiles share each Wq B-fragment ----
  v8f qacc1[8];  // m-tile 1 parked in registers until xs is dead
  {
    const _Float16* wq = Wqh + (size_t)h * DH * CC;
#pragma unroll
    for (int n = 0; n < 8; ++n) {
      v8f q0 = zero8(), q1 = zero8();
      const _Float16* wrow = wq + (size_t)(n * 16 + rowA) * CC;
      for (int kt = 0; kt < 16; ++kt) {
        int k0 = kt * 32;
        v16h bb = frag_h(wrow + k0 + koffA, wrow + k0 + 16 + koffA);
        v16h a0 = frag_h(&xs[rowA][k0 + koffA], &xs[rowA][k0 + 16 + koffA]);
        v16h a1 =
            frag_h(&xs[16 + rowA][k0 + koffA], &xs[16 + rowA][k0 + 16 + koffA]);
        q0 = wmma16(a0, bb, q0);
        q1 = wmma16(a1, bb, q1);
      }
      int cl = n * 16 + rowA;
      float bias = Wq_b[h * DH + cl];
#pragma unroll
      for (int r = 0; r < 8; ++r) {
        qs0[h][r + halfsel * 8][cl] = (_Float16)(q0[r] + bias);
        qacc1[n][r] = q1[r] + bias;
      }
    }
  }
  __syncthreads();  // all waves done reading xs -> its region is reusable

  // park m-tile-1 Q into LDS (aliases dead xs region); frees qacc1
#pragma unroll
  for (int n = 0; n < 8; ++n) {
    int cl = n * 16 + rowA;
#pragma unroll
    for (int r = 0; r < 8; ++r)
      qs1[h][r + halfsel * 8][cl] = (_Float16)qacc1[n][r];
  }

  // mask bits (same for both m-tiles)
  int okm[6];
  {
    const int* mrow = amask + b * LL;
#pragma unroll
    for (int lt = 0; lt < 6; ++lt) {
      int col = lt * 16 + rowA;
      okm[lt] = (col < LL) ? (mrow[col] != 0 ? 1 : 0) : 0;
    }
  }

  const _Float16* kb = Kbuf + (size_t)b * LP * CC + h * DH;
  const _Float16* vtb = Vtbuf + ((size_t)b * CC + h * DH) * LP;
#pragma unroll
  for (int lt = 0; lt < 6; ++lt)
    __builtin_prefetch(kb + (size_t)(lt * 16 + rowA) * CC, 0, 3);
#pragma unroll
  for (int n = 0; n < 8; ++n)
    __builtin_prefetch(vtb + (size_t)(n * 16 + rowA) * LP, 0, 3);

  // ---- fused scores: each K B-fragment feeds both m-tiles ----
  v8f sacc[2][6];
#pragma unroll
  for (int m = 0; m < 2; ++m)
#pragma unroll
    for (int lt = 0; lt < 6; ++lt) sacc[m][lt] = zero8();
  for (int kt = 0; kt < 4; ++kt) {
    int k0 = kt * 32;
    v16h a0 = frag_h(&qs0[h][rowA][k0 + koffA], &qs0[h][rowA][k0 + 16 + koffA]);
    v16h a1 = frag_h(&qs1[h][rowA][k0 + koffA], &qs1[h][rowA][k0 + 16 + koffA]);
#pragma unroll
    for (int lt = 0; lt < 6; ++lt) {
      const _Float16* krow = kb + (size_t)(lt * 16 + rowA) * CC + k0;
      v16h bb = frag_h(krow + koffA, krow + 16 + koffA);
      sacc[0][lt] = wmma16(a0, bb, sacc[0][lt]);
      sacc[1][lt] = wmma16(a1, bb, sacc[1][lt]);
    }
  }

  // ---- masked softmax over L (per m-tile) ----
  const float scale = 0.088388347648318447f;  // 128^-0.5
#pragma unroll
  for (int m = 0; m < 2; ++m) {
    _Float16(*psm)[16][LP + 8] = (m == 0) ? ps0 : ps1;
#pragma unroll
    for (int r = 0; r < 8; ++r) {
      float v[6];
      float mx = -1e30f;
#pragma unroll
      for (int lt = 0; lt < 6; ++lt) {
        float t = okm[lt] ? sacc[m][lt][r] * scale : -1e30f;
        v[lt] = t;
        mx = fmaxf(mx, t);
      }
#pragma unroll
      for (int off = 1; off < 16; off <<= 1)
        mx = fmaxf(mx, __shfl_xor(mx, off, 32));
      float sum = 0.0f;
#pragma unroll
      for (int lt = 0; lt < 6; ++lt) {
        float e = okm[lt] ? __expf(v[lt] - mx) : 0.0f;
        v[lt] = e;
        sum += e;
      }
#pragma unroll
      for (int off = 1; off < 16; off <<= 1)
        sum += __shfl_xor(sum, off, 32);
      float inv = 1.0f / sum;
#pragma unroll
      for (int lt = 0; lt < 6; ++lt)
        psm[h][r + halfsel * 8][lt * 16 + rowA] = (_Float16)(v[lt] * inv);
    }
  }

  // ---- fused PV: each V B-fragment feeds both m-tiles ----
  v8f oacc[2][8];
#pragma unroll
  for (int n = 0; n < 8; ++n) {
    oacc[0][n] = zero8();
    oacc[1][n] = zero8();
    const _Float16* vrow = vtb + (size_t)(n * 16 + rowA) * LP;
#pragma unroll
    for (int kt = 0; kt < 3; ++kt) {
      int k0 = kt * 32;
      v16h a0 =
          frag_h(&ps0[h][rowA][k0 + koffA], &ps0[h][rowA][k0 + 16 + koffA]);
      v16h a1 =
          frag_h(&ps1[h][rowA][k0 + koffA], &ps1[h][rowA][k0 + 16 + koffA]);
      v16h bb = frag_h(vrow + k0 + koffA, vrow + k0 + 16 + koffA);
      oacc[0][n] = wmma16(a0, bb, oacc[0][n]);
      oacc[1][n] = wmma16(a1, bb, oacc[1][n]);
    }
  }

  // ---- deterministic group stats: per 16-query-tile partials ----
#pragma unroll
  for (int m = 0; m < 2; ++m) {
    float gsum[2] = {0.0f, 0.0f}, gsq[2] = {0.0f, 0.0f};
#pragma unroll
    for (int n = 0; n < 8; ++n) {
      float s_ = 0.0f, sq = 0.0f;
#pragma unroll
      for (int r = 0; r < 8; ++r) {
        float vv = oacc[m][n][r];
        s_ += vv;
        sq += vv * vv;
      }
      int g = n >> 2;  // 64 channels per group = 4 n-tiles
      gsum[g] += s_;
      gsq[g] += sq;
    }
#pragma unroll
    for (int off = 1; off < 32; off <<= 1) {
      gsum[0] += __shfl_xor(gsum[0], off, 32);
      gsq[0] += __shfl_xor(gsq[0], off, 32);
      gsum[1] += __shfl_xor(gsum[1], off, 32);
      gsq[1] += __shfl_xor(gsq[1], off, 32);
    }
    if (lane == 0) {
      int st16 = stp * 2 + m;
      size_t p0 = ((size_t)(b * 8 + h * 2 + 0) * 256 + st16) * 2;
      size_t p1 = ((size_t)(b * 8 + h * 2 + 1) * 256 + st16) * 2;
      part[p0] = gsum[0];
      part[p0 + 1] = gsq[0];
      part[p1] = gsum[1];
      part[p1 + 1] = gsq[1];
    }
  }

  // ---- coalesced writeout: stage tiles in LDS (aliases everything) ----
  __syncthreads();  // all waves done with qs0/qs1/ps0/ps1
#pragma unroll
  for (int m = 0; m < 2; ++m) {
#pragma unroll
    for (int n = 0; n < 8; ++n) {
      int cl = n * 16 + rowA;
#pragma unroll
      for (int r = 0; r < 8; ++r) outT[h][cl][halfsel * 8 + r] = oacc[m][n][r];
    }
    // 4 lanes cover one channel's 16 floats (64B run) -> dense b128 stores
    float* obase = out + ((size_t)b * CC + (size_t)h * DH) * SS + s0 + m * 16;
#pragma unroll
    for (int pass = 0; pass < 16; ++pass) {
      int cl = pass * 8 + (lane >> 2);
      int sc = (lane & 3) * 4;
      float4 vv = *(const float4*)&outT[h][cl][sc];
      *(float4*)(obase + (size_t)cl * SS + sc) = vv;
    }
  }
}

// ---------------------------------------------------------------------------
// stat: deterministic fixed-order reduction of per-tile partials
// grid = B*GROUPS blocks
// ---------------------------------------------------------------------------
__global__ __launch_bounds__(256) void stat_kernel(const float* __restrict__ part,
                                                   float* __restrict__ stats) {
  __shared__ float ssum[256];
  __shared__ float ssq[256];
  int bg = blockIdx.x;
  int t = threadIdx.x;
  ssum[t] = part[((size_t)bg * 256 + t) * 2 + 0];
  ssq[t] = part[((size_t)bg * 256 + t) * 2 + 1];
  __syncthreads();
  for (int o = 128; o > 0; o >>= 1) {
    if (t < o) {
      ssum[t] += ssum[t + o];
      ssq[t] += ssq[t + o];
    }
    __syncthreads();
  }
  if (t == 0) {
    stats[bg * 2 + 0] = ssum[0];
    stats[bg * 2 + 1] = ssq[0];
  }
}

// ---------------------------------------------------------------------------
// gn: in-place GroupNorm over d_out ([B][C][H][W] == [b][c][s])
// ---------------------------------------------------------------------------
__global__ __launch_bounds__(256) void gn_kernel(const float* __restrict__ stats,
                                                 const float* __restrict__ gsc,
                                                 const float* __restrict__ gbi,
                                                 float* __restrict__ out) {
  size_t i = ((size_t)blockIdx.x * 256 + threadIdx.x) * 4;
  int b = (int)(i >> 21);          // 512*4096 = 2^21
  int c = (int)((i >> 12) & 511);  // 4096 = 2^12
  int g = c >> 6;
  const float N = 64.0f * 4096.0f;
  float sum = stats[(b * 8 + g) * 2 + 0];
  float sq = stats[(b * 8 + g) * 2 + 1];
  float mean = sum / N;
  float var = sq / N - mean * mean;
  float rinv = rsqrtf(var + 1e-5f);
  float sc = gsc[c] * rinv;
  float bi = gbi[c] - mean * sc;
  float4 vv = *(const float4*)(out + i);
  vv.x = vv.x * sc + bi;
  vv.y = vv.y * sc + bi;
  vv.z = vv.z * sc + bi;
  vv.w = vv.w * sc + bi;
  *(float4*)(out + i) = vv;
}

// ---------------------------------------------------------------------------
extern "C" void kernel_launch(void* const* d_in, const int* in_sizes, int n_in,
                              void* d_out, int out_size, void* d_ws,
                              size_t ws_size, hipStream_t stream) {
  const float* x = (const float*)d_in[0];
  const float* te = (const float*)d_in[1];
  const int* am = (const int*)d_in[2];
  const float* Wq_w = (const float*)d_in[3];
  const float* Wq_b = (const float*)d_in[4];
  const float* Wk_w = (const float*)d_in[5];
  const float* Wk_b = (const float*)d_in[6];
  const float* Wv_w = (const float*)d_in[7];
  const float* Wv_b = (const float*)d_in[8];
  const float* gsc = (const float*)d_in[9];
  const float* gbi = (const float*)d_in[10];
  float* out = (float*)d_out;

  char* ws = (char*)d_ws;
  _Float16* Kbuf = (_Float16*)(ws);            // 8*96*512*2 = 786432 B
  _Float16* Vtbuf = (_Float16*)(ws + 786432);  // 786432 B
  _Float16* Wqh = (_Float16*)(ws + 1572864);   // 512*512*2 = 524288 B
  float* part = (float*)(ws + 2097152);        // 64*256*2 floats = 131072 B
  float* stats = (float*)(ws + 2228224);       // 128 floats

  prep_kernel<<<1024, 256, 0, stream>>>(Wq_w, Wqh);
  kv_kernel<<<BB * 6 * 32, 32, 0, stream>>>(te, Wk_w, Wk_b, Wv_w, Wv_b, Kbuf,
                                            Vtbuf);
  attn_kernel<<<BB * 128, 128, 0, stream>>>(x, am, Kbuf, Vtbuf, Wqh, Wq_b, out,
                                            part);
  stat_kernel<<<64, 256, 0, stream>>>(part, stats);
  gn_kernel<<<16384, 256, 0, stream>>>(stats, gsc, gbi, out);
}
